// Model_5_TraceGenerator_34153579938560
// MI455X (gfx1250) — matrix operand order repaired and checked by reading, not compiled, not run on hardware
//
#include <hip/hip_runtime.h>

// ---------------------------------------------------------------------------
// Fused 2-layer LSTM for MI455X (gfx1250, wave32, WMMA).
//
// Per wave: 16 batch samples. Recurrence per timestep:
//   G[16x128] = A[16x32] x B[32x128] + P    (8x v_wmma_f32_16x16x32_f16)
// A rows = h[t-1] (K=0..29), K=30 = AVG[t], K=31 = unused (B row 31 = 0).
// B columns permuted gate-major: tiles (0,1)=i, (2,3)=f, (4,5)=g, (6,7)=o,
// so the elementwise cell update needs NO cross-lane traffic.
// P = feat-projection + b1, computed once via 8 prologue WMMAs.
// h D-layout -> A-layout transpose goes through per-wave private LDS
// (same-wave DS is in-order; no barriers in the steady-state loop).
// Weight matrices are staged block-cooperatively into LDS with branchless
// clamped loads, then pulled into registers as contiguous ds_load_b128s.
// ---------------------------------------------------------------------------

typedef __attribute__((ext_vector_type(16))) _Float16 v16h;
typedef __attribute__((ext_vector_type(8)))  _Float16 v8h;
typedef __attribute__((ext_vector_type(8)))  float    v8f;

__constant__ float AVG_C[120] = {
  0.0256f,0.0823f,0.1157f,0.1315f,0.1366f,0.1369f,0.1347f,0.1308f,0.1259f,0.1205f,
  0.1146f,0.1086f,0.1028f,0.0970f,0.0913f,0.0858f,0.0805f,0.0756f,0.0708f,0.0664f,
  0.0623f,0.0584f,0.0549f,0.0515f,0.0485f,0.0456f,0.0429f,0.0404f,0.0381f,0.0360f,
  0.0340f,0.0321f,0.0304f,0.0287f,0.0272f,0.0258f,0.0245f,0.0233f,0.0222f,0.0211f,
  0.0201f,0.0191f,0.0182f,0.0173f,0.0165f,0.0158f,0.0150f,0.0143f,0.0137f,0.0130f,
  0.0125f,0.0119f,0.0114f,0.0108f,0.0104f,0.0099f,0.0095f,0.0091f,0.0087f,0.0083f,
  0.0080f,0.0077f,0.0074f,0.0071f,0.0068f,0.0065f,0.0062f,0.0060f,0.0058f,0.0055f,
  0.0053f,0.0050f,0.0049f,0.0047f,0.0045f,0.0044f,0.0042f,0.0040f,0.0039f,0.0038f,
  0.0036f,0.0034f,0.0033f,0.0032f,0.0031f,0.0030f,0.0029f,0.0028f,0.0027f,0.0026f,
  0.0025f,0.0024f,0.0023f,0.0022f,0.0021f,0.0021f,0.0020f,0.0019f,0.0018f,0.0018f,
  0.0017f,0.0017f,0.0016f,0.0016f,0.0015f,0.0015f,0.0014f,0.0014f,0.0013f,0.0013f,
  0.0013f,0.0012f,0.0012f,0.0011f,0.0011f,0.0011f,0.0010f,0.0010f,0.0010f,0.0009f };

__device__ __forceinline__ float fsig(float x) {
  // sigmoid(x) = 1 / (1 + 2^(-x*log2e))
  return __builtin_amdgcn_rcpf(1.0f + __builtin_amdgcn_exp2f(-1.4426950408889634f * x));
}
__device__ __forceinline__ float ftanh(float x) {
  // tanh(x) = 1 - 2/(2^(2x*log2e) + 1) ; saturates correctly at +-inf
  float e = __builtin_amdgcn_exp2f(2.8853900817779268f * x);
  return 1.0f - 2.0f * __builtin_amdgcn_rcpf(e + 1.0f);
}

__global__ __launch_bounds__(256)
void lstm2_fused_kernel(const float* __restrict__ feat,   // [N,12]
                        const float* __restrict__ wih1,   // [120,13]
                        const float* __restrict__ whh1,   // [120,30]
                        const float* __restrict__ b1,     // [120]
                        const float* __restrict__ wih2,   // [4,30]
                        const float* __restrict__ whh2,   // [4,1]
                        const float* __restrict__ b2,     // [4]
                        float* __restrict__ out,          // [N,120]
                        int N)
{
  __shared__ __align__(16) _Float16 BextS[128 * 32];         // recurrent B, [c][K]
  __shared__ __align__(16) _Float16 BihS [128 * 32];         // input-proj B, [c][K]
  __shared__ __align__(16) _Float16 hbuf_all[8 * 16 * 32];   // per-wave h transpose buffer
  __shared__ __align__(16) float    obuf_all[8 * 16 * 120];  // per-wave output staging

  const int tid  = threadIdx.x;
  const int lane = tid & 31;
  const int wid  = tid >> 5;
  const int hi   = lane >> 4;          // 0 = lanes 0-15, 1 = lanes 16-31
  const int lm   = lane & 15;
  const long base = ((long)blockIdx.x * 8 + wid) * 16;   // first sample of this wave

  // ===== block-cooperative weight staging (branchless clamped loads) =======
  // thread t fills column c = t>>1, K range [(t&1)*16, +16)
  {
    int c    = tid >> 1;
    int k0   = (tid & 1) << 4;
    int gate = c >> 5;
    int u    = c & 31;                       // output unit; u>=30 is pad
    bool um  = (u < 30);
    int grow = gate * 30 + (um ? u : 0);
    float vw13 = wih1[grow * 13 + 12];       // AVG-channel weight
    float vb1  = b1[grow];                   // combined bias
    _Float16* bc = BextS + c * 32;
    _Float16* ic = BihS  + c * 32;
    #pragma unroll
    for (int kk = 0; kk < 16; ++kk) {
      int K = k0 + kk;
      // recurrent B: K<30 -> whh1 ; K==30 -> w13 ; K==31 -> 0
      float vwhh = whh1[grow * 30 + (K < 30 ? K : 0)];
      float vb = (K < 30) ? vwhh : (K == 30 ? vw13 : 0.0f);
      bc[K] = (_Float16)(um ? vb : 0.0f);
      // input-proj B: K<12 -> wih1 ; K==31 -> b1 ; else 0
      float vwih = wih1[grow * 13 + (K < 12 ? K : 0)];
      float vi = (K < 12) ? vwih : (K == 31 ? vb1 : 0.0f);
      ic[K] = (_Float16)(um ? vi : 0.0f);
    }
  }
  __syncthreads();
  if (base >= N) return;

  _Float16* hbuf = hbuf_all + wid * (16 * 32);
  float*    obuf = obuf_all + wid * (16 * 120);

  // zero the h buffer (h[-1] = 0); slots 30/31 of each row stay 0 forever
  {
    unsigned int* hz = (unsigned int*)hbuf;   // 256 dwords per wave
    #pragma unroll
    for (int i = 0; i < 8; ++i) hz[lane + 32 * i] = 0u;
  }

  // --- pull recurrent B tiles into registers: 2x ds_load_b128 per tile -----
  // B f16 layout: lane holds column c = 16j + lm; element e<8 -> K = e + 8*hi,
  // e>=8 -> K = 16 + 8*hi + (e-8), two 8-half runs in the [c][K] LDS layout.
  v16h B[8];
  #pragma unroll
  for (int j = 0; j < 8; ++j) {
    const v8h* p = (const v8h*)(BextS + (16 * j + lm) * 32 + 8 * hi);
    v8h x0 = p[0];
    v8h x1 = p[2];
    #pragma unroll
    for (int e = 0; e < 8; ++e) { B[j][e] = x0[e]; B[j][8 + e] = x1[e]; }
  }

  // --- P = features @ wih1[:, :12]^T + b1, via WMMA, kept in D layout ------
  // A_feat: row = lane%16; element e -> K map (A 16-bit layout):
  //   lanes 0-15 : e<8 -> K=e,   e>=8 -> K=e+8   (K in 0..7, 16..23)
  //   lanes 16-31: e<8 -> K=e+8, e>=8 -> K=e+16  (K in 8..15, 24..31)
  v8f P[8];
  {
    v16h af;
    long rowA = base + lm; if (rowA >= N) rowA = N - 1;
    const float* fr = feat + rowA * 12;
    #pragma unroll
    for (int e = 0; e < 16; ++e) {
      int K = (hi == 0) ? (e < 8 ? e : e + 8) : (e < 8 ? e + 8 : e + 16);
      float v = fr[K < 12 ? K : 0];                       // unconditional load
      v = (K < 12) ? v : (K == 31 ? 1.0f : 0.0f);         // bias-row multiplier
      af[e] = (_Float16)v;
    }
    #pragma unroll
    for (int j = 0; j < 8; ++j) {
      const v8h* p = (const v8h*)(BihS + (16 * j + lm) * 32 + 8 * hi);
      v8h x0 = p[0];
      v8h x1 = p[2];
      v16h bi;
      #pragma unroll
      for (int e = 0; e < 8; ++e) { bi[e] = x0[e]; bi[8 + e] = x1[e]; }
      v8f z = {};
      P[j] = __builtin_amdgcn_wmma_f32_16x16x32_f16(false, af, false, bi,
                                                    (short)0, z, false, false);
    }
  }

  // --- LSTM2 per-lane weights in A-element order (units >= 30 -> 0) --------
  v16h w2h[4];
  #pragma unroll
  for (int g = 0; g < 4; ++g) {
    #pragma unroll
    for (int e = 0; e < 16; ++e) {
      int K = (hi == 0) ? (e < 8 ? e : e + 8) : (e < 8 ? e + 8 : e + 16);
      float v = wih2[g * 30 + (K < 30 ? K : 0)];
      w2h[g][e] = (_Float16)((K < 30) ? v : 0.0f);
    }
  }
  float whh2r[4], b2r[4];
  #pragma unroll
  for (int g = 0; g < 4; ++g) { whh2r[g] = whh2[g]; b2r[g] = b2[g]; }

  v8f c1[2]; c1[0] = (v8f){}; c1[1] = (v8f){};   // layer-1 cell state (D layout)
  float h2 = 0.0f, c2 = 0.0f;                    // layer-2 state (per row)

  const int rowb  = lm * 32;                     // hbuf row base (halfs)
  const int aoff0 = hi ? 8 : 0;                  // K base of first 8 elements
  const int aoff1 = hi ? 24 : 16;                // K base of second 8 elements

  for (int t = 0; t < 120; ++t) {
    // ---- build A (h[t-1] + AVG[t] slot) from hbuf -------------------------
    v16h a;
    {
      v8h lo = *(const v8h*)(hbuf + rowb + aoff0);
      v8h hh = *(const v8h*)(hbuf + rowb + aoff1);
      #pragma unroll
      for (int e = 0; e < 8; ++e) { a[e] = lo[e]; a[8 + e] = hh[e]; }
      _Float16 avgh = (_Float16)AVG_C[t];
      a[14] = hi ? avgh : a[14];                 // K=30 slot (lanes 16-31)
      a[15] = hi ? (_Float16)0.0f : a[15];       // K=31 slot (B row 31 = 0)
    }

    // ---- 8 WMMAs + elementwise gates, two unit-halves ---------------------
    #pragma unroll
    for (int p = 0; p < 2; ++p) {
      v8f gi = __builtin_amdgcn_wmma_f32_16x16x32_f16(false, a, false, B[0 + p], (short)0, P[0 + p], false, false);
      v8f gf = __builtin_amdgcn_wmma_f32_16x16x32_f16(false, a, false, B[2 + p], (short)0, P[2 + p], false, false);
      v8f gg = __builtin_amdgcn_wmma_f32_16x16x32_f16(false, a, false, B[4 + p], (short)0, P[4 + p], false, false);
      v8f go = __builtin_amdgcn_wmma_f32_16x16x32_f16(false, a, false, B[6 + p], (short)0, P[6 + p], false, false);
      #pragma unroll
      for (int r = 0; r < 8; ++r) {
        float cn = fsig(gf[r]) * c1[p][r] + fsig(gi[r]) * ftanh(gg[r]);
        float hn = fsig(go[r]) * ftanh(cn);
        c1[p][r] = cn;
        // D layout -> hbuf[row][unit]: row = r + 8*hi, unit = 16p + lane%16
        hbuf[(r + 8 * hi) * 32 + (16 * p + lm)] = (_Float16)hn;
      }
    }

    // ---- fused LSTM2 (hidden = 1): dot(h[t], wih2) + scalar recurrence ----
    v8h l2 = *(const v8h*)(hbuf + rowb + aoff0);   // h[t] in A layout
    v8h h2v = *(const v8h*)(hbuf + rowb + aoff1);
    float acc0 = 0.f, acc1 = 0.f, acc2 = 0.f, acc3 = 0.f;
    #pragma unroll
    for (int e = 0; e < 8; ++e) {
      float x0 = (float)l2[e], x1 = (float)h2v[e];
      acc0 += x0 * (float)w2h[0][e] + x1 * (float)w2h[0][8 + e];
      acc1 += x0 * (float)w2h[1][e] + x1 * (float)w2h[1][8 + e];
      acc2 += x0 * (float)w2h[2][e] + x1 * (float)w2h[2][8 + e];
      acc3 += x0 * (float)w2h[3][e] + x1 * (float)w2h[3][8 + e];
    }
    // lanes l and l+16 hold complementary unit halves of row l -> one xor-16
    acc0 += __shfl_xor(acc0, 16, 32);
    acc1 += __shfl_xor(acc1, 16, 32);
    acc2 += __shfl_xor(acc2, 16, 32);
    acc3 += __shfl_xor(acc3, 16, 32);
    float zi = acc0 + b2r[0] + h2 * whh2r[0];
    float zf = acc1 + b2r[1] + h2 * whh2r[1];
    float zg = acc2 + b2r[2] + h2 * whh2r[2];
    float zo = acc3 + b2r[3] + h2 * whh2r[3];
    c2 = fsig(zf) * c2 + fsig(zi) * ftanh(zg);
    h2 = fsig(zo) * ftanh(c2);
    if (!hi) obuf[lm * 120 + t] = h2;
  }

  // ---- coalesced flush: 16 rows x 120 f32 contiguous in out ---------------
  if (base + 16 <= (long)N) {
    const float4* src = (const float4*)obuf;
    float4* dst = (float4*)(out + base * 120);
    #pragma unroll
    for (int i = 0; i < 15; ++i) dst[lane + 32 * i] = src[lane + 32 * i];
  } else {
    long limit = (long)N * 120;
    for (int i = lane; i < 16 * 120; i += 32) {
      long idx = base * 120 + i;
      if (idx < limit) out[idx] = obuf[i];
    }
  }
}

extern "C" void kernel_launch(void* const* d_in, const int* in_sizes, int n_in,
                              void* d_out, int out_size, void* d_ws, size_t ws_size,
                              hipStream_t stream) {
  (void)n_in; (void)out_size; (void)d_ws; (void)ws_size;
  const float* feat = (const float*)d_in[0];   // [N,12]
  const float* wih1 = (const float*)d_in[1];   // [120,13]
  const float* whh1 = (const float*)d_in[2];   // [120,30]
  const float* b1   = (const float*)d_in[3];   // [120]
  const float* wih2 = (const float*)d_in[4];   // [4,30]
  const float* whh2 = (const float*)d_in[5];   // [4,1]
  const float* b2   = (const float*)d_in[6];   // [4]
  float* out = (float*)d_out;                  // [N,120]

  int N = in_sizes[0] / 12;                    // 32768
  int blocks = (N + 127) / 128;                // 128 samples per 256-thread block
  lstm2_fused_kernel<<<blocks, 256, 0, stream>>>(feat, wih1, whh1, b1,
                                                 wih2, whh2, b2, out, N);
}
